// RWKV_ChannelMix_63144609185889
// MI455X (gfx1250) — matrix operand order repaired and checked by reading, not compiled, not run on hardware
//
#include <hip/hip_runtime.h>

// ---------------- CDNA5 WMMA types ----------------
typedef __attribute__((ext_vector_type(16))) __bf16 v16bf;
typedef __attribute__((ext_vector_type(8)))  __bf16 v8bf;
typedef __attribute__((ext_vector_type(8)))  float  v8f;

__device__ __forceinline__ unsigned short f32_to_bf16(float f) {
  union { float f; unsigned int u; } v; v.f = f;
  unsigned int u = v.u;
  u += 0x7FFFu + ((u >> 16) & 1u);   // round-to-nearest-even
  return (unsigned short)(u >> 16);
}

__device__ __forceinline__ v8f wmma_bf16(v16bf a, v16bf b, v8f c) {
  // D = A(16x32) * B(32x16) + C, fp32 accumulate
  return __builtin_amdgcn_wmma_f32_16x16x32_bf16(
      /*neg_a=*/false, a, /*neg_b=*/false, b,
      /*c_mod=*/(short)0, c, /*reuse_a=*/false, /*reuse_b=*/false);
}

// A-fragment: 16-bit A 16x32.  Lane L: row M=L%16, K = k + (L/16)*16 .. +15
// -> one contiguous 32B load.
__device__ __forceinline__ v16bf load_a_frag(const __bf16* __restrict__ X,
                                             int row, int ld, int k, int lane) {
  const __bf16* p = X + (size_t)row * ld + k + ((lane >> 4) << 4);
  return *(const v16bf*)p;
}

// B-fragment: 16-bit B 32x16.  Lane L: col N=L%16, K = k + (L/16)*16 .. +15
// -> one contiguous 32B load (row N of W, since B = W^T).
__device__ __forceinline__ v16bf load_b_frag(const __bf16* __restrict__ W,
                                             int col, int ld, int k, int lane) {
  const __bf16* p = W + (size_t)col * ld + k + ((lane >> 4) << 4);
  return *(const v16bf*)p;
}

// ---------------- Phase 1: token shift + bf16 convert + new_state ----------------
__global__ __launch_bounds__(256)
void prep_shift(const float* __restrict__ x,
                const float* __restrict__ mu_r, const float* __restrict__ mu_k,
                unsigned short* __restrict__ xr, unsigned short* __restrict__ xk,
                float* __restrict__ state, int T, int C, int BT) {
  int cq  = C >> 2;
  int gid = blockIdx.x * 256 + threadIdx.x;
  int row = gid / cq;
  if (row >= BT) return;
  int c = (gid - row * cq) << 2;

  const float4 xv = *(const float4*)(x + (size_t)row * C + c);
  float4 xp = make_float4(0.f, 0.f, 0.f, 0.f);
  if ((row % T) != 0) xp = *(const float4*)(x + (size_t)(row - 1) * C + c);
  const float4 mr = *(const float4*)(mu_r + c);
  const float4 mk = *(const float4*)(mu_k + c);

  ushort4 orr, okk;
  orr.x = f32_to_bf16(mr.x * xv.x + (1.f - mr.x) * xp.x);
  orr.y = f32_to_bf16(mr.y * xv.y + (1.f - mr.y) * xp.y);
  orr.z = f32_to_bf16(mr.z * xv.z + (1.f - mr.z) * xp.z);
  orr.w = f32_to_bf16(mr.w * xv.w + (1.f - mr.w) * xp.w);
  okk.x = f32_to_bf16(mk.x * xv.x + (1.f - mk.x) * xp.x);
  okk.y = f32_to_bf16(mk.y * xv.y + (1.f - mk.y) * xp.y);
  okk.z = f32_to_bf16(mk.z * xv.z + (1.f - mk.z) * xp.z);
  okk.w = f32_to_bf16(mk.w * xv.w + (1.f - mk.w) * xp.w);
  *(ushort4*)(xr + (size_t)row * C + c) = orr;
  *(ushort4*)(xk + (size_t)row * C + c) = okk;

  if ((row % T) == T - 1) {
    int b = row / T;
    *(float4*)(state + (size_t)b * C + c) = xv;
  }
}

// ---------------- Phase 1b: weights -> bf16 ----------------
__global__ __launch_bounds__(256)
void convert_w(const float* __restrict__ w0, const float* __restrict__ w1,
               const float* __restrict__ w2,
               unsigned short* __restrict__ o0, unsigned short* __restrict__ o1,
               unsigned short* __restrict__ o2, int n) {
  int gid = (blockIdx.x * 256 + threadIdx.x) << 2;
  if (gid >= n) return;
  int m = blockIdx.y;
  const float* src = (m == 0) ? w0 : ((m == 1) ? w1 : w2);
  unsigned short* dst = (m == 0) ? o0 : ((m == 1) ? o1 : o2);
  float4 v = *(const float4*)(src + gid);
  ushort4 o;
  o.x = f32_to_bf16(v.x); o.y = f32_to_bf16(v.y);
  o.z = f32_to_bf16(v.z); o.w = f32_to_bf16(v.w);
  *(ushort4*)(dst + gid) = o;
}

// ---------------- Phase 2: kact = relu(xk @ Wk^T)^2 (bf16 out) ----------------
// Macro tile 128(M) x 256(N); 8 waves 2(M) x 4(N); wave tile 64x64.
// Per k-step: 16 WMMA per 16 b128 loads -> 512 B of cache traffic per WMMA.
__global__ __launch_bounds__(256)
void gemm_kact(const unsigned short* __restrict__ Xk_,
               const unsigned short* __restrict__ Wk_,
               unsigned short* __restrict__ Kact, int C) {
  const __bf16* X = (const __bf16*)Xk_;
  const __bf16* W = (const __bf16*)Wk_;
  const int lane = threadIdx.x & 31;
  const int wid  = threadIdx.x >> 5;
  const int m0 = blockIdx.y * 128 + (wid & 1) * 64;
  const int n0 = blockIdx.x * 256 + (wid >> 1) * 64;
  const int rA = m0 + (lane & 15);
  const int rB = n0 + (lane & 15);

  v8f acc[4][4] = {};
  for (int k = 0; k < C; k += 32) {
    v16bf A[4], Bf[4];
#pragma unroll
    for (int mi = 0; mi < 4; ++mi) A[mi]  = load_a_frag(X, rA + mi * 16, C, k, lane);
#pragma unroll
    for (int ni = 0; ni < 4; ++ni) Bf[ni] = load_b_frag(W, rB + ni * 16, C, k, lane);
#pragma unroll
    for (int mi = 0; mi < 4; ++mi)
#pragma unroll
      for (int ni = 0; ni < 4; ++ni)
        acc[mi][ni] = wmma_bf16(A[mi], Bf[ni], acc[mi][ni]);
  }

  const int mOff = (lane >> 4) << 3;  // row = r + 8*(lane/16)
  const int nCol = lane & 15;
#pragma unroll
  for (int mi = 0; mi < 4; ++mi)
#pragma unroll
    for (int ni = 0; ni < 4; ++ni)
#pragma unroll
      for (int r = 0; r < 8; ++r) {
        float v = acc[mi][ni][r];
        v = v > 0.f ? v * v : 0.f;
        Kact[(size_t)(m0 + mi * 16 + r + mOff) * C + (n0 + ni * 16 + nCol)] =
            f32_to_bf16(v);
      }
}

// ------- Phase 3: out = sigmoid(xr @ Wr^T) * (kact @ Wv^T), fp32 out -------
// Macro tile 64(M) x 256(N); 8 waves 2(M) x 4(N); wave tile 32x64.
// Two sequential K-loops share fragment registers; two accumulator sets live.
__global__ __launch_bounds__(256)
void gemm_out(const unsigned short* __restrict__ Xr_,
              const unsigned short* __restrict__ Wr_,
              const unsigned short* __restrict__ Ka_,
              const unsigned short* __restrict__ Wv_,
              float* __restrict__ Out, int C) {
  const __bf16* Xr = (const __bf16*)Xr_;
  const __bf16* Wr = (const __bf16*)Wr_;
  const __bf16* Ka = (const __bf16*)Ka_;
  const __bf16* Wv = (const __bf16*)Wv_;
  const int lane = threadIdx.x & 31;
  const int wid  = threadIdx.x >> 5;
  const int m0 = blockIdx.y * 64  + (wid & 1) * 32;
  const int n0 = blockIdx.x * 256 + (wid >> 1) * 64;
  const int rA = m0 + (lane & 15);
  const int rB = n0 + (lane & 15);

  // GEMM 1: r = xr @ Wr^T
  v8f accR[2][4] = {};
  for (int k = 0; k < C; k += 32) {
    v16bf A[2], Bf[4];
#pragma unroll
    for (int mi = 0; mi < 2; ++mi) A[mi]  = load_a_frag(Xr, rA + mi * 16, C, k, lane);
#pragma unroll
    for (int ni = 0; ni < 4; ++ni) Bf[ni] = load_b_frag(Wr, rB + ni * 16, C, k, lane);
#pragma unroll
    for (int mi = 0; mi < 2; ++mi)
#pragma unroll
      for (int ni = 0; ni < 4; ++ni)
        accR[mi][ni] = wmma_bf16(A[mi], Bf[ni], accR[mi][ni]);
  }
#pragma unroll
  for (int mi = 0; mi < 2; ++mi)
#pragma unroll
    for (int ni = 0; ni < 4; ++ni)
#pragma unroll
      for (int r = 0; r < 8; ++r)
        accR[mi][ni][r] = 1.f / (1.f + __expf(-accR[mi][ni][r]));

  // GEMM 2: v = kact @ Wv^T
  v8f accV[2][4] = {};
  for (int k = 0; k < C; k += 32) {
    v16bf A[2], Bf[4];
#pragma unroll
    for (int mi = 0; mi < 2; ++mi) A[mi]  = load_a_frag(Ka, rA + mi * 16, C, k, lane);
#pragma unroll
    for (int ni = 0; ni < 4; ++ni) Bf[ni] = load_b_frag(Wv, rB + ni * 16, C, k, lane);
#pragma unroll
    for (int mi = 0; mi < 2; ++mi)
#pragma unroll
      for (int ni = 0; ni < 4; ++ni)
        accV[mi][ni] = wmma_bf16(A[mi], Bf[ni], accV[mi][ni]);
  }

  const int mOff = (lane >> 4) << 3;
  const int nCol = lane & 15;
#pragma unroll
  for (int mi = 0; mi < 2; ++mi)
#pragma unroll
    for (int ni = 0; ni < 4; ++ni)
#pragma unroll
      for (int r = 0; r < 8; ++r)
        Out[(size_t)(m0 + mi * 16 + r + mOff) * C + (n0 + ni * 16 + nCol)] =
            accR[mi][ni][r] * accV[mi][ni][r];
}

// ---------------- host-side launcher ----------------
extern "C" void kernel_launch(void* const* d_in, const int* in_sizes, int n_in,
                              void* d_out, int out_size, void* d_ws, size_t ws_size,
                              hipStream_t stream) {
  const float* x    = (const float*)d_in[0];
  const float* W_r  = (const float*)d_in[1];
  const float* W_k  = (const float*)d_in[2];
  const float* W_v  = (const float*)d_in[3];
  const float* mu_r = (const float*)d_in[4];
  const float* mu_k = (const float*)d_in[5];

  const int C  = in_sizes[4];            // 1024
  const int BT = in_sizes[0] / C;        // 16384
  const int B  = out_size / C - BT;      // 8
  const int T  = BT / B;                 // 2048

  float* out   = (float*)d_out;
  float* state = out + (size_t)BT * C;

  unsigned short* xr = (unsigned short*)d_ws;
  unsigned short* xk = xr + (size_t)BT * C;
  unsigned short* ka = xk + (size_t)BT * C;
  unsigned short* wr = ka + (size_t)BT * C;
  unsigned short* wk = wr + (size_t)C * C;
  unsigned short* wv = wk + (size_t)C * C;

  {
    long total = (long)BT * (C / 4);
    prep_shift<<<dim3((unsigned)((total + 255) / 256)), 256, 0, stream>>>(
        x, mu_r, mu_k, xr, xk, state, T, C, BT);
  }
  {
    int n = C * C;
    convert_w<<<dim3((n / 4 + 255) / 256, 3), 256, 0, stream>>>(
        W_r, W_k, W_v, wr, wk, wv, n);
  }
  {
    dim3 g(C / 256, BT / 128);
    gemm_kact<<<g, 256, 0, stream>>>(xk, wk, ka, C);
  }
  {
    dim3 g(C / 256, BT / 64);
    gemm_out<<<g, 256, 0, stream>>>(xr, wr, ka, wv, out, C);
  }
}
